// DoubleAtten_90701119357068
// MI455X (gfx1250) — compile-verified, matrix operand order checked by reading, not run on hardware
//
#include <hip/hip_runtime.h>

typedef __attribute__((ext_vector_type(16))) __bf16 v16bf;
typedef __attribute__((ext_vector_type(8)))  float  v8f;
typedef __attribute__((ext_vector_type(4)))  unsigned int u32x4;
typedef __attribute__((ext_vector_type(8)))  int    i32x8;
typedef __attribute__((ext_vector_type(4)))  int    i32x4;
typedef __attribute__((ext_vector_type(4)))  float  f32x4;

#define DEVI static __device__ __forceinline__

namespace {

constexpr int kB  = 8;
constexpr int kC  = 512;
constexpr int kHW = 4096;

DEVI unsigned short f32_to_bf16_rn(float f) {
  unsigned u = __builtin_bit_cast(unsigned, f);
  u += 0x7FFFu + ((u >> 16) & 1u);
  return (unsigned short)(u >> 16);
}
DEVI float bf16_to_f32(unsigned short h) {
  return __builtin_bit_cast(float, (unsigned)h << 16);
}

union FragAB  { v16bf v; u32x4 q[2]; };
union FragAcc { v8f   v; float f[8]; };

// ---------------------------------------------------------------------------
// TDM: async-load a 128-row x 32-col (bf16) 2D tile into LDS.
// D# group0: count=1, lds_addr, 57b global_addr, type=2 ("image").
// D# group1: data_size=2B; pad_enable with pad_interval=16 DWORDs (one 64B
//            row) and pad_amount=4 DWORDs (16B) so rows land on the padded
//            80B LDS pitch; tile_dim0=32 elems, tile_dim1=128 rows;
//            tensor_dim0_stride = row pitch in elements.
// 6-arg builtin form (clang-23 / therock headers): groups 2,3 as i32x4,
// trailing i32x8 group, then cache policy.
// ---------------------------------------------------------------------------
DEVI void tdm_load_tile_128x32_bf16(unsigned lds_byte_off,
                                    const unsigned short* gsrc,
                                    unsigned ld_elems,
                                    unsigned tensor_d0,
                                    unsigned tensor_d1) {
  unsigned long long ga = (unsigned long long)gsrc;
  u32x4 g0;
  g0[0] = 1u;                                                  // count=1
  g0[1] = lds_byte_off;                                        // LDS dest
  g0[2] = (unsigned)ga;                                        // global lo
  g0[3] = (unsigned)((ga >> 32) & 0x01FFFFFFu) | (2u << 30);   // hi + type=2
  i32x8 g1;
  g1[0] = (int)((1u << 16) |                                   // data_size=2B
                (1u << 20) |                                   // pad_enable
                (3u << 22) |                                   // pad_interval=16DW
                (3u << 25));                                   // pad_amount=4DW
  g1[1] = (int)((tensor_d0 & 0xFFFFu) << 16);                  // dim0[15:0]
  g1[2] = (int)((tensor_d0 >> 16) | ((tensor_d1 & 0xFFFFu) << 16));
  g1[3] = (int)((tensor_d1 >> 16) | (32u << 16));              // tile_dim0=32
  g1[4] = (int)128u;                                           // tile_dim1=128
  g1[5] = (int)ld_elems;                                       // stride0 lo
  g1[6] = 0;
  g1[7] = 0;
  i32x4 z4 = {0, 0, 0, 0};
  i32x8 z8 = {0, 0, 0, 0, 0, 0, 0, 0};
  __builtin_amdgcn_tensor_load_to_lds(g0, g1, z4, z4, z8, 0);
}

// Manual transposed staging: tile[r][k] = src[(k0+k)*ld + base0 + r]
// (source rows contiguous along r; 2x b128 global loads + b16 LDS scatter).
DEVI void stage_kmajor_transpose(unsigned short (*tile)[40],
                                 const unsigned short* src, int ld,
                                 int k0, int base0, int t) {
  const int k  = t >> 3;          // 0..31
  const int rb = (t & 7) * 16;    // 0,16,...,112
  const unsigned short* s = src + (long long)(k0 + k) * ld + base0 + rb;
  union { u32x4 q; unsigned short h[8]; } u0, u1;
  u0.q = *(const u32x4*)(s);
  u1.q = *(const u32x4*)(s + 8);
  #pragma unroll
  for (int i = 0; i < 8; ++i) tile[rb + i][k] = u0.h[i];
  #pragma unroll
  for (int i = 0; i < 8; ++i) tile[rb + 8 + i][k] = u1.h[i];
}

} // namespace

// ---------------------------------------------------------------------------
// Tiled bf16 GEMM with f32 accumulation:
//   C[m,n] = scale * sum_k Agemm[m,k]*Bgemm[k,n] (+ bias[m])
//   Agemm[m,k] = TRA ? A[k*lda+m] : A[m*lda+k]
//   Bgemm[k,n] = TRB ? B[n*ldb+k] : B[k*ldb+n]
// Block tile 128x128, K-step 32 (one v_wmma_f32_16x16x32_bf16 per subtile).
// A-tile uses TDM when !TRA (rows K-contiguous); B-tile uses TDM when TRB.
// ---------------------------------------------------------------------------
template<bool TRA, bool TRB, bool HAS_BIAS, bool OUT_F32>
__global__ __launch_bounds__(256)
void gemm_bf16_wmma(const unsigned short* __restrict__ A, long long sA, int lda,
                    const unsigned short* __restrict__ B, long long sB, int ldb,
                    const float* __restrict__ bias,
                    void* __restrict__ Cout, long long sC, int ldc,
                    int M, int N, int K, float scale) {
  constexpr bool TDM_A = !TRA;
  constexpr bool TDM_B = TRB;
  (void)M; (void)N;

  __shared__ __align__(16) unsigned short As[2][128][40]; // padded 80B pitch
  __shared__ __align__(16) unsigned short Bs[2][128][40];

  const int t    = (int)threadIdx.x;
  const int lane = t & 31;
  const int wv   = t >> 5;       // 8 waves
  const int wm   = wv >> 2;      // 0..1  -> 64-row strip
  const int wn   = wv & 3;       // 0..3  -> 32-col strip
  const int hi   = lane >> 4;    // K-chunk select per ISA bf16 layout
  const int l15  = lane & 15;    // row/col within 16x16 tile
  const int m0   = (int)blockIdx.y * 128;
  const int n0   = (int)blockIdx.x * 128;

  const unsigned short* Ab = A + (long long)blockIdx.z * sA;
  const unsigned short* Bb = B + (long long)blockIdx.z * sB;

  FragAcc acc[4][2];
  #pragma unroll
  for (int i = 0; i < 4; ++i)
    #pragma unroll
    for (int j = 0; j < 2; ++j)
      #pragma unroll
      for (int r = 0; r < 8; ++r) acc[i][j].f[r] = 0.0f;

  const int nk = K >> 5;

  auto stage = [&](int kb, int buf) {
    const int k0 = kb << 5;
    if constexpr (TDM_A) {
      if (t < 32)  // wave 0 issues the DMA (EXEC ignored by TDM)
        tdm_load_tile_128x32_bf16(
            (unsigned)(unsigned long long)&As[buf][0][0],
            Ab + (long long)m0 * lda + k0,
            (unsigned)lda, (unsigned)lda, 1u << 20);
    } else {
      stage_kmajor_transpose(As[buf], Ab, lda, k0, m0, t);
    }
    if constexpr (TDM_B) {
      if (t < 32)
        tdm_load_tile_128x32_bf16(
            (unsigned)(unsigned long long)&Bs[buf][0][0],
            Bb + (long long)n0 * ldb + k0,
            (unsigned)ldb, (unsigned)ldb, 1u << 20);
    } else {
      stage_kmajor_transpose(Bs[buf], Bb, ldb, k0, n0, t);
    }
  };

  stage(0, 0);

  for (int kb = 0; kb < nk; ++kb) {
    const int buf = kb & 1;
    if constexpr (TDM_A || TDM_B) {
      if (t < 32) __builtin_amdgcn_s_wait_tensorcnt(0);
    }
    __syncthreads();                         // tile[buf] visible to all waves
    if (kb + 1 < nk) stage(kb + 1, buf ^ 1); // prefetch next K-slab

    // Fragment loads per ISA bf16 A/B layout: lane l15 = row (A) / col (B);
    // per-lane K chunks [kc..kc+7] and [kc+16..kc+23], kc = hi*8.
    FragAB fa[4], fb[2];
    const int kc = hi * 8;
    #pragma unroll
    for (int i = 0; i < 4; ++i) {
      const unsigned short* pr = &As[buf][wm * 64 + i * 16 + l15][kc];
      fa[i].q[0] = *(const u32x4*)pr;
      fa[i].q[1] = *(const u32x4*)(pr + 16);
    }
    #pragma unroll
    for (int j = 0; j < 2; ++j) {
      const unsigned short* pr = &Bs[buf][wn * 32 + j * 16 + l15][kc];
      fb[j].q[0] = *(const u32x4*)pr;
      fb[j].q[1] = *(const u32x4*)(pr + 16);
    }

    #pragma unroll
    for (int i = 0; i < 4; ++i)
      #pragma unroll
      for (int j = 0; j < 2; ++j)
        acc[i][j].v = __builtin_amdgcn_wmma_f32_16x16x32_bf16(
            false, fa[i].v, false, fb[j].v, (short)0, acc[i][j].v,
            false, false);
  }

  // Epilogue: C/D layout — VGPR r: lanes0-15 -> M=r, lanes16-31 -> M=r+8.
  #pragma unroll
  for (int i = 0; i < 4; ++i) {
    #pragma unroll
    for (int j = 0; j < 2; ++j) {
      const int rowb = m0 + wm * 64 + i * 16 + hi * 8;
      const int col  = n0 + wn * 32 + j * 16 + l15;
      #pragma unroll
      for (int r = 0; r < 8; ++r) {
        const int row = rowb + r;
        float v = acc[i][j].f[r] * scale;
        if constexpr (HAS_BIAS) v += bias[row];
        if constexpr (OUT_F32) {
          float* C = (float*)Cout + (long long)blockIdx.z * sC;
          C[(long long)row * ldc + col] = v;
        } else {
          unsigned short* C = (unsigned short*)Cout + (long long)blockIdx.z * sC;
          C[(long long)row * ldc + col] = f32_to_bf16_rn(v);
        }
      }
    }
  }
}

// In-place row softmax over bf16 rows of length 4096 (f32 math).
__global__ __launch_bounds__(256)
void softmax_row_bf16(unsigned short* __restrict__ data, int rowlen) {
  unsigned short* p = data + (long long)blockIdx.x * rowlen;
  const int t = (int)threadIdx.x;
  float v[16];
  float lmax = -3.402823466e+38f;
  #pragma unroll
  for (int i = 0; i < 16; ++i) {
    v[i] = bf16_to_f32(p[t * 16 + i]);
    lmax = fmaxf(lmax, v[i]);
  }
  __shared__ float red[256];
  red[t] = lmax;
  __syncthreads();
  for (int s = 128; s > 0; s >>= 1) {
    if (t < s) red[t] = fmaxf(red[t], red[t + s]);
    __syncthreads();
  }
  const float m = red[0];
  __syncthreads();
  float lsum = 0.0f;
  #pragma unroll
  for (int i = 0; i < 16; ++i) {
    v[i] = __expf(v[i] - m);
    lsum += v[i];
  }
  red[t] = lsum;
  __syncthreads();
  for (int s = 128; s > 0; s >>= 1) {
    if (t < s) red[t] += red[t + s];
    __syncthreads();
  }
  const float inv = 1.0f / red[0];
  #pragma unroll
  for (int i = 0; i < 16; ++i) p[t * 16 + i] = f32_to_bf16_rn(v[i] * inv);
}

// 4-wide f32 -> bf16 conversion.
__global__ __launch_bounds__(256)
void cvt_f32_to_bf16(const float* __restrict__ in,
                     unsigned short* __restrict__ out, int n) {
  const int i4 = ((int)blockIdx.x * 256 + (int)threadIdx.x) * 4;
  if (i4 + 3 < n) {
    f32x4 f = *(const f32x4*)(in + i4);
    unsigned long long pk =
        (unsigned long long)f32_to_bf16_rn(f[0]) |
        ((unsigned long long)f32_to_bf16_rn(f[1]) << 16) |
        ((unsigned long long)f32_to_bf16_rn(f[2]) << 32) |
        ((unsigned long long)f32_to_bf16_rn(f[3]) << 48);
    *(unsigned long long*)(out + i4) = pk;
  }
}

extern "C" void kernel_launch(void* const* d_in, const int* in_sizes, int n_in,
                              void* d_out, int out_size, void* d_ws, size_t ws_size,
                              hipStream_t stream) {
  (void)in_sizes; (void)n_in; (void)out_size; (void)ws_size;
  const float* x  = (const float*)d_in[0];
  const float* wA = (const float*)d_in[1];
  const float* bA = (const float*)d_in[2];
  const float* wB = (const float*)d_in[3];
  const float* bB = (const float*)d_in[4];
  const float* wV = (const float*)d_in[5];
  const float* bV = (const float*)d_in[6];

  const long long NX = (long long)kB * kC * kHW;   // 16,777,216
  const long long NW = (long long)kC * kC;         // 262,144

  // Workspace carve (bf16 halves all intermediate traffic): ~140 MB total.
  unsigned short* xb   = (unsigned short*)d_ws;
  unsigned short* Alin = xb + NX;
  unsigned short* Bm   = Alin + NX;
  unsigned short* Vm   = Bm + NX;
  unsigned short* gd   = Vm + NX;                  // kB*kC*kC
  unsigned short* wAb  = gd + (long long)kB * kC * kC;
  unsigned short* wBb  = wAb + NW;
  unsigned short* wVb  = wBb + NW;

  dim3 blk(256);

  cvt_f32_to_bf16<<<(unsigned)(NX / 4 / 256), blk, 0, stream>>>(x, xb, (int)NX);
  cvt_f32_to_bf16<<<(unsigned)(NW / 4 / 256), blk, 0, stream>>>(wA, wAb, (int)NW);
  cvt_f32_to_bf16<<<(unsigned)(NW / 4 / 256), blk, 0, stream>>>(wB, wBb, (int)NW);
  cvt_f32_to_bf16<<<(unsigned)(NW / 4 / 256), blk, 0, stream>>>(wV, wVb, (int)NW);

  // Stage 1: three 1x1 convs, per batch  W[512,512] @ x_b[512,4096] + bias.
  dim3 g1(kHW / 128, kC / 128, kB);
  gemm_bf16_wmma<false, false, true, false><<<g1, blk, 0, stream>>>(
      wAb, 0LL, kC, xb, (long long)kC * kHW, kHW, bA,
      Alin, (long long)kC * kHW, kHW, kC, kHW, kC, 1.0f);
  gemm_bf16_wmma<false, false, true, false><<<g1, blk, 0, stream>>>(
      wBb, 0LL, kC, xb, (long long)kC * kHW, kHW, bB,
      Bm, (long long)kC * kHW, kHW, kC, kHW, kC, 1.0f);
  gemm_bf16_wmma<false, false, true, false><<<g1, blk, 0, stream>>>(
      wVb, 0LL, kC, xb, (long long)kC * kHW, kHW, bV,
      Vm, (long long)kC * kHW, kHW, kC, kHW, kC, 1.0f);

  // Stage 2: softmax over HW (in place).
  softmax_row_bf16<<<kB * kC, blk, 0, stream>>>(Bm, kHW);
  softmax_row_bf16<<<kB * kC, blk, 0, stream>>>(Vm, kHW);

  // Stage 3: gd = (Bm @ A^T) / HW   [512 x 512], K = 4096 (TDM on A and B).
  dim3 g3(kC / 128, kC / 128, kB);
  gemm_bf16_wmma<false, true, false, false><<<g3, blk, 0, stream>>>(
      Bm, (long long)kC * kHW, kHW, Alin, (long long)kC * kHW, kHW, nullptr,
      gd, (long long)kC * kC, kC, kC, kC, kHW, 1.0f / (float)kHW);

  // Stage 4: out = gd^T @ av   [512 x 4096], K = 512, f32 output.
  dim3 g4(kHW / 128, kC / 128, kB);
  gemm_bf16_wmma<true, false, false, true><<<g4, blk, 0, stream>>>(
      gd, (long long)kC * kC, kC, Vm, (long long)kC * kHW, kHW, nullptr,
      d_out, (long long)kC * kHW, kHW, kC, kHW, kC, 1.0f);
}
